// LSTM_66846870995327
// MI455X (gfx1250) — matrix operand order repaired and checked by reading, not compiled, not run on hardware
//
#include <hip/hip_runtime.h>
#include <hip/hip_bf16.h>
#include <stdint.h>

// MI455X / gfx1250 LSTM.
// - Input projections hoisted out of the time loop (big fp16 WMMA GEMMs).
// - Serial path: per-timestep kernel; W_hh slab (shared by all 4 waves of a
//   block) staged into LDS via CDNA5 async global->LDS copies (ASYNCcnt),
//   double-buffered so DMA overlaps WMMA. W_hh (8MB fp16) stays L2-resident.

typedef __attribute__((ext_vector_type(16))) _Float16 v16h;
typedef __attribute__((ext_vector_type(8)))  _Float16 v8h;
typedef __attribute__((ext_vector_type(8)))  float    v8f;
typedef __attribute__((ext_vector_type(4)))  int      v4i;

constexpr int Bn = 64, Tn = 512, Dn = 512, Hn = 1024;
constexpr int Gn = 4 * Hn;      // 4096 gate columns
constexpr int RS = 72;          // LDS row pitch in halves (144B) to spread banks

#if defined(__has_builtin)
#  if __has_builtin(__builtin_amdgcn_global_load_async_to_lds_b128)
#    define USE_ASYNC_LDS 1
#  endif
#endif
#ifndef USE_ASYNC_LDS
#  define USE_ASYNC_LDS 0
#endif

#define AS1 __attribute__((address_space(1)))
#define AS3 __attribute__((address_space(3)))

#if USE_ASYNC_LDS
#  if __has_builtin(__builtin_amdgcn_s_wait_asynccnt)
#    define WAIT_ASYNC(n) __builtin_amdgcn_s_wait_asynccnt(n)
#  else
#    define WAIT_ASYNC(n) asm volatile("s_wait_asynccnt " #n ::: "memory")
#  endif
#else
#  define WAIT_ASYNC(n)
#endif

// ---- WMMA fragment loaders (wave32, contiguous k order used for both A and B, 16-bit A 16x32 / B 32x16) ----
// A (16x32, MxK): lanes 0-15 hold row M=lane, K=0..15;
// lanes 16-31 hold row M=lane-16, K=16..31.
__device__ __forceinline__ v16h load_a_frag(const _Float16* A, size_t lda,
                                            int m0, int k0, int lane) {
  const int r  = lane & 15;
  const int hf = lane >> 4;
  const _Float16* p = A + (size_t)(m0 + r) * lda + k0;
  v8h lo = *(const v8h*)(p + hf * 16);
  v8h hi = *(const v8h*)(p + hf * 16 + 8);
  v16h o;
#pragma unroll
  for (int i = 0; i < 8; ++i) { o[i] = lo[i]; o[i + 8] = hi[i]; }
  return o;
}

// B (32x16, KxN) with W stored row-major [N,K] (B = W^T): lane c holds column
// N=c; lanes 0-15 K=0..15, lanes 16-31 K=16..31. One contiguous 32B read.
__device__ __forceinline__ v16h load_b_frag(const _Float16* W, size_t ldw,
                                            int n0, int k0, int lane) {
  const int c  = lane & 15;
  const int hf = lane >> 4;
  const _Float16* p = W + (size_t)(n0 + c) * ldw + k0 + hf * 16;
  v8h lo = *(const v8h*)(p);
  v8h hi = *(const v8h*)(p + 8);
  v16h o;
#pragma unroll
  for (int i = 0; i < 8; ++i) { o[i] = lo[i]; o[i + 8] = hi[i]; }
  return o;
}

// ---- fp32 -> fp16 conversion ----
__global__ void f32_to_f16(const float* __restrict__ in, _Float16* __restrict__ out, size_t n) {
  size_t i = (size_t)blockIdx.x * blockDim.x + threadIdx.x;
  size_t stride = (size_t)gridDim.x * blockDim.x;
  for (; i < n; i += stride) out[i] = (_Float16)in[i];
}

// ---- Big input-projection GEMM: xg[m,n] = sum_k A[m,k]*W[n,k] + b0[n]+b1[n] ----
__global__ __launch_bounds__(128) void gemm_xg(
    const _Float16* __restrict__ A, int lda,
    const _Float16* __restrict__ W, int ldw,
    const float* __restrict__ b0, const float* __restrict__ b1,
    _Float16* __restrict__ xg, int K)
{
  const int lane = threadIdx.x & 31;
  const int wave = threadIdx.x >> 5;
  const int m0 = blockIdx.y * 64 + wave * 16;
  const int n0 = blockIdx.x * 64;
  v8f acc[4] = {};
  for (int k0 = 0; k0 < K; k0 += 32) {
    v16h a = load_a_frag(A, (size_t)lda, m0, k0, lane);
#pragma unroll
    for (int tn = 0; tn < 4; ++tn) {
      v16h b = load_b_frag(W, (size_t)ldw, n0 + 16 * tn, k0, lane);
      acc[tn] = __builtin_amdgcn_wmma_f32_16x16x32_f16(
          false, a, false, b, (short)0, acc[tn], false, false);
    }
  }
#pragma unroll
  for (int tn = 0; tn < 4; ++tn) {
    const int n = n0 + 16 * tn + (lane & 15);
    const float bias = b0[n] + b1[n];
#pragma unroll
    for (int r = 0; r < 8; ++r) {
      const int m = m0 + (lane >> 4) * 8 + r;
      xg[(size_t)m * Gn + n] = (_Float16)(acc[tn][r] + bias);
    }
  }
}

// ---- One recurrent timestep, gates fused, LDS-staged W_hh ----
// grid = Hn/16 blocks, block = 128 (4 waves). Block owns hidden units
// j..j+15 for all 4 gates (64 W_hh rows); waves split the batch (m0=wave*16).
// The 64-row K-slab is staged into LDS in 64-half chunks (8KB), double
// buffered: async DMA of chunk c+1 overlaps WMMA on chunk c.
__global__ __launch_bounds__(128) void lstm_step(
    const _Float16* __restrict__ hprev, int hstride,
    const _Float16* __restrict__ Whh,
    const _Float16* __restrict__ xg, int t,
    float* __restrict__ c, _Float16* __restrict__ hseq)
{
  __shared__ _Float16 wlds[2][64 * RS];   // 2 x 9KB
  const int tid  = threadIdx.x;
  const int lane = tid & 31;
  const int wave = tid >> 5;
  const int m0 = wave * 16;
  const int j  = blockIdx.x * 16;

  // Stage one 64-row x 64-half chunk: 512 x 16B transfers over 128 threads.
  auto stage = [&](int buf, int k0) {
#pragma unroll
    for (int i = 0; i < 4; ++i) {
      const int q   = tid + i * 128;
      const int row = q >> 3;             // 0..63 = gate*16 + col
      const int sub = q & 7;              // 16B unit within the 128B row chunk
      const int g   = row >> 4;
      const int col = row & 15;
      const _Float16* gp = Whh + ((size_t)(g * Hn + j + col) * Hn + k0) + sub * 8;
      _Float16* lp = &wlds[buf][row * RS + sub * 8];
#if USE_ASYNC_LDS
      __builtin_amdgcn_global_load_async_to_lds_b128(
          (AS1 v4i*)(uintptr_t)gp,
          (AS3 v4i*)(uint32_t)(uintptr_t)lp, 0, 0);
#else
      *(v8h*)lp = *(const v8h*)gp;        // global_load_b128 + ds_store_b128
#endif
    }
  };

  v8f acc[4] = {};
  stage(0, 0);
  int buf = 0;
  constexpr int NCHUNK = Hn / 64;         // 16
  for (int chunk = 0; chunk < NCHUNK; ++chunk) {
    if (chunk + 1 < NCHUNK) {
      stage(buf ^ 1, (chunk + 1) * 64);
      WAIT_ASYNC(4);                      // current buffer landed, next in flight
    } else {
      WAIT_ASYNC(0);
    }
    __syncthreads();
#pragma unroll
    for (int kk = 0; kk < 2; ++kk) {      // two 16x16x32 K-steps per chunk
      const int k0 = chunk * 64 + kk * 32;
      v16h a = load_a_frag(hprev, (size_t)hstride, m0, k0, lane);
#pragma unroll
      for (int g = 0; g < 4; ++g) {
        const _Float16* bp = &wlds[buf][(g * 16 + (lane & 15)) * RS +
                                        kk * 32 + (lane >> 4) * 16];
        v8h lo = *(const v8h*)bp;         // ds_load_b128
        v8h hi = *(const v8h*)(bp + 8);   // ds_load_b128
        v16h b;
#pragma unroll
        for (int i = 0; i < 8; ++i) { b[i] = lo[i]; b[i + 8] = hi[i]; }
        acc[g] = __builtin_amdgcn_wmma_f32_16x16x32_f16(
            false, a, false, b, (short)0, acc[g], false, false);
      }
    }
    __syncthreads();                      // done reading buf before restaging it
    buf ^= 1;
  }

  const int n = j + (lane & 15);
#pragma unroll
  for (int r = 0; r < 8; ++r) {
    const int m = m0 + (lane >> 4) * 8 + r;          // batch index
    const size_t row  = (size_t)(m * Tn + t);
    const size_t xrow = row * Gn;
    float iv = (float)xg[xrow + n]           + acc[0][r];
    float fv = (float)xg[xrow + Hn + n]      + acc[1][r];
    float gv = (float)xg[xrow + 2 * Hn + n]  + acc[2][r];
    float ov = (float)xg[xrow + 3 * Hn + n]  + acc[3][r];
    iv = 1.f / (1.f + __expf(-iv));
    fv = 1.f / (1.f + __expf(-fv));
    gv = tanhf(gv);
    ov = 1.f / (1.f + __expf(-ov));
    const int ci = m * Hn + n;
    const float cn = fv * c[ci] + iv * gv;
    c[ci] = cn;
    hseq[row * Hn + n] = (_Float16)(ov * tanhf(cn));
  }
}

// ---- Final FC: out[b] = h_last[b,:] . Wfc + bfc ----
__global__ __launch_bounds__(256) void fc_kernel(
    const _Float16* __restrict__ hseq,
    const float* __restrict__ Wfc, const float* __restrict__ bfc,
    float* __restrict__ out)
{
  __shared__ float red[256];
  const int b = blockIdx.x;
  const _Float16* h = hseq + ((size_t)(b * Tn + (Tn - 1))) * Hn;
  float s = 0.f;
  for (int i = threadIdx.x; i < Hn; i += 256) s += (float)h[i] * Wfc[i];
  red[threadIdx.x] = s;
  __syncthreads();
  for (int off = 128; off > 0; off >>= 1) {
    if (threadIdx.x < off) red[threadIdx.x] += red[threadIdx.x + off];
    __syncthreads();
  }
  if (threadIdx.x == 0) out[b] = red[0] + bfc[0];
}

extern "C" void kernel_launch(void* const* d_in, const int* in_sizes, int n_in,
                              void* d_out, int out_size, void* d_ws, size_t ws_size,
                              hipStream_t stream) {
  (void)in_sizes; (void)n_in; (void)out_size; (void)ws_size;
  const float* x    = (const float*)d_in[0];
  const float* Wih0 = (const float*)d_in[1];
  const float* Whh0 = (const float*)d_in[2];
  const float* bih0 = (const float*)d_in[3];
  const float* bhh0 = (const float*)d_in[4];
  const float* Wih1 = (const float*)d_in[5];
  const float* Whh1 = (const float*)d_in[6];
  const float* bih1 = (const float*)d_in[7];
  const float* bhh1 = (const float*)d_in[8];
  const float* Wfc  = (const float*)d_in[9];
  const float* bfc  = (const float*)d_in[10];
  float* out = (float*)d_out;

  // Workspace carve (~466 MB; dominated by the 268 MB fp16 xg buffer).
  uint8_t* p = (uint8_t*)d_ws;
  auto take = [&](size_t bytes) -> uint8_t* {
    uint8_t* r = p; p += (bytes + 255) & ~(size_t)255; return r;
  };
  _Float16* x16   = (_Float16*)take((size_t)Bn * Tn * Dn * 2);
  _Float16* wih0  = (_Float16*)take((size_t)Gn * Dn * 2);
  _Float16* whh0  = (_Float16*)take((size_t)Gn * Hn * 2);
  _Float16* wih1  = (_Float16*)take((size_t)Gn * Hn * 2);
  _Float16* whh1  = (_Float16*)take((size_t)Gn * Hn * 2);
  _Float16* xg    = (_Float16*)take((size_t)Bn * Tn * Gn * 2);
  _Float16* hseq0 = (_Float16*)take((size_t)Bn * Tn * Hn * 2);
  _Float16* hseq1 = (_Float16*)take((size_t)Bn * Tn * Hn * 2);
  _Float16* hzero = (_Float16*)take((size_t)Bn * Hn * 2);
  float*    cbuf  = (float*)   take((size_t)Bn * Hn * 4);

  // fp32 -> fp16 staging
  f32_to_f16<<<2048, 256, 0, stream>>>(x,    x16,  (size_t)Bn * Tn * Dn);
  f32_to_f16<<<1024, 256, 0, stream>>>(Wih0, wih0, (size_t)Gn * Dn);
  f32_to_f16<<<1024, 256, 0, stream>>>(Whh0, whh0, (size_t)Gn * Hn);
  f32_to_f16<<<1024, 256, 0, stream>>>(Wih1, wih1, (size_t)Gn * Hn);
  f32_to_f16<<<1024, 256, 0, stream>>>(Whh1, whh1, (size_t)Gn * Hn);
  (void)hipMemsetAsync(hzero, 0, (size_t)Bn * Hn * 2, stream);

  const dim3 ggrid(Gn / 64, (Bn * Tn) / 64);  // 64 x 512 blocks

  // ---- Layer 0 ----
  gemm_xg<<<ggrid, 128, 0, stream>>>(x16, Dn, wih0, Dn, bih0, bhh0, xg, Dn);
  (void)hipMemsetAsync(cbuf, 0, (size_t)Bn * Hn * 4, stream);
  for (int t = 0; t < Tn; ++t) {
    const _Float16* hp = (t == 0) ? hzero : (hseq0 + (size_t)(t - 1) * Hn);
    const int hs = (t == 0) ? Hn : Tn * Hn;
    lstm_step<<<Hn / 16, 128, 0, stream>>>(hp, hs, whh0, xg, t, cbuf, hseq0);
  }

  // ---- Layer 1 ----
  gemm_xg<<<ggrid, 128, 0, stream>>>(hseq0, Hn, wih1, Hn, bih1, bhh1, xg, Hn);
  (void)hipMemsetAsync(cbuf, 0, (size_t)Bn * Hn * 4, stream);
  for (int t = 0; t < Tn; ++t) {
    const _Float16* hp = (t == 0) ? hzero : (hseq1 + (size_t)(t - 1) * Hn);
    const int hs = (t == 0) ? Hn : Tn * Hn;
    lstm_step<<<Hn / 16, 128, 0, stream>>>(hp, hs, whh1, xg, t, cbuf, hseq1);
  }

  // ---- Final FC ----
  fc_kernel<<<Bn, 256, 0, stream>>>(hseq1, Wfc, bfc, out);
}
